// RNN_23742579212415
// MI455X (gfx1250) — matrix operand order repaired and checked by reading, not compiled, not run on hardware
//
#include <hip/hip_runtime.h>

typedef __attribute__((ext_vector_type(16))) _Float16 v16h;
typedef __attribute__((ext_vector_type(2)))  __fp16   h2;
typedef __attribute__((ext_vector_type(8)))  float    v8f;

namespace {

constexpr int kT = 512;   // time steps
constexpr int kD = 16;    // input features
// H = 32 hidden, batch tile = 16 per workgroup

// Scale folded into the weights/biases so the WMMA accumulator is already
// y = 2*log2(e) * (W_ih x + b + W_hh h); then tanh(x) = 1 - 2/(2^y + 1).
constexpr float kS = 2.8853900817779268f;   // 2*log2(e)

union AB16 {              // one f16 WMMA A- or B-fragment: 8 dwords per lane
    v16h v;
    _Float16 h[16];
    unsigned u[8];
};

__device__ __forceinline__ v8f wmma_f32_f16(v16h a, v16h b, v8f c) {
    // D = A(16x32 f16) * B(32x16 f16) + C(16x16 f32)
    return __builtin_amdgcn_wmma_f32_16x16x32_f16(false, a, false, b,
                                                  (short)0, c, false, false);
}

__device__ __forceinline__ unsigned pack2(float a, float b) {
    // single v_cvt_pk_rtz_f16_f32
    union { h2 v; unsigned u; } p;
    p.v = __builtin_amdgcn_cvt_pkrtz(a, b);
    return p.u;
}

__device__ __forceinline__ float tanh_from_scaled(float y) {
    // y = 2*log2(e)*x; tanh(x) = 1 - 2/(2^y + 1); exact +-1 saturation.
    const float e = __builtin_amdgcn_exp2f(y);        // v_exp_f32
    const float r = __builtin_amdgcn_rcpf(e + 1.0f);  // v_rcp_f32
    return __builtin_fmaf(-2.0f, r, 1.0f);            // v_fma_f32
}

// Build one 16x32 f16 A-fragment from row-major f32 W (M rows mbase..mbase+15,
// row stride ldw), pre-scaled by `scale`. Columns k >= kmax zero-padded
// (used for W_ih0, K=16). A layout: lanes 0-15 hold M=lane,
// K in 0..15; lanes 16-31 hold M=lane-16, K in 16..31;
// dword j packs (k, k+1).
__device__ __forceinline__ v16h load_A_tile(const float* __restrict__ W,
                                            int mbase, int ldw, int kmax,
                                            int lane, float scale) {
    const int m = mbase + (lane & 15);
    const int khalf = (lane & 16) ? 16 : 0;
    AB16 a;
#pragma unroll
    for (int j = 0; j < 8; ++j) {
        const int kb = 2 * j;
#pragma unroll
        for (int s = 0; s < 2; ++s) {
            const int k = kb + khalf + s;
            a.h[2 * j + s] =
                (_Float16)((k < kmax) ? (scale * W[m * ldw + k]) : 0.0f);
        }
    }
    return a.v;
}

// Accumulators acc0 (hidden rows 0..15) / acc1 (rows 16..31) are in the D
// layout (lane = batch column n, dword v = hidden row), pre-scaled by kS.
// Apply tanh and repack into the B layout for the next step's WMMA:
//   B lanes 0-15: K=0..15 (tile0), lanes 16-31: K=16..31 (tile1).
// Lo lanes only need the partner's tile0 pairs, hi lanes only tile1 pairs,
// so one shfl_xor per dword pair suffices (send what the partner needs).
__device__ __forceinline__ void acc_to_B(const v8f& acc0, const v8f& acc1,
                                         float* t0, float* t1,
                                         unsigned* breg, int lane) {
    const bool lo = (lane < 16);
#pragma unroll
    for (int v = 0; v < 8; ++v) {
        t0[v] = tanh_from_scaled(acc0[v]);
        t1[v] = tanh_from_scaled(acc1[v]);
    }
#pragma unroll
    for (int j = 0; j < 4; ++j) {
        const unsigned p0 = pack2(t0[2 * j], t0[2 * j + 1]);
        const unsigned p1 = pack2(t1[2 * j], t1[2 * j + 1]);
        const unsigned snd = lo ? p1 : p0;                    // partner's need
        const unsigned rcv = (unsigned)__shfl_xor((int)snd, 16, 32);
        breg[j]     = lo ? p0 : rcv;  // K=2j..2j+1 (lo) / 16+2j..17+2j (hi)
        breg[j + 4] = lo ? rcv : p1;  // K=8+2j..9+2j (lo) / 24+2j..25+2j (hi)
    }
}

// Load x_t^T tile (K=feature 0..15, N=batch=lane) as B-fragment dwords.
// Lanes 16-31 cover K=16..31 which multiply the zero-padded half of the
// W_ih0 A-fragment — keep them zero.
__device__ __forceinline__ void load_x_tile(const float* __restrict__ xrow,
                                            int t, unsigned* xr, bool lo) {
    if (lo) {
        const float4* p = (const float4*)(xrow + (size_t)t * kD);
#pragma unroll
        for (int q = 0; q < 4; ++q) {
            const float4 f = p[q];
            xr[2 * q]     = pack2(f.x, f.y);
            xr[2 * q + 1] = pack2(f.z, f.w);
        }
    } else {
#pragma unroll
        for (int j = 0; j < 8; ++j) xr[j] = 0u;
    }
}

} // namespace

__global__ __launch_bounds__(64)
void rnn2_wmma_fused(const float* __restrict__ x,
                     const float* __restrict__ W_ih0,
                     const float* __restrict__ W_hh0,
                     const float* __restrict__ b_ih0,
                     const float* __restrict__ b_hh0,
                     const float* __restrict__ W_ih1,
                     const float* __restrict__ W_hh1,
                     const float* __restrict__ b_ih1,
                     const float* __restrict__ b_hh1,
                     const float* __restrict__ W_fc,
                     const float* __restrict__ b_fc,
                     float* __restrict__ out)
{
    // h0_t handoff in B-layout, double buffered; padded row (9) vs 64 banks
    __shared__ unsigned hbuf[2][32][9];

    const int lane = threadIdx.x & 31;
    const int wave = threadIdx.x >> 5;
    const int b0   = blockIdx.x * 16;       // batch tile base
    const bool lo  = (lane < 16);
    const int moff = lo ? 0 : 8;

    if (wave == 0) {
        // ================= layer-0 producer wave =================
        const v16h a_ih0 = load_A_tile(W_ih0,  0, 16, 16, lane, kS); // K pad
        const v16h a_ih1 = load_A_tile(W_ih0, 16, 16, 16, lane, kS);
        const v16h a_hh0 = load_A_tile(W_hh0,  0, 32, 32, lane, kS);
        const v16h a_hh1 = load_A_tile(W_hh0, 16, 32, 32, lane, kS);

        v8f cb0, cb1;                 // scaled bias folded into WMMA C operand
#pragma unroll
        for (int v = 0; v < 8; ++v) {
            const int m = v + moff;
            cb0[v] = kS * (b_ih0[m]      + b_hh0[m]);
            cb1[v] = kS * (b_ih0[m + 16] + b_hh0[m + 16]);
        }

        unsigned hreg[8];
#pragma unroll
        for (int j = 0; j < 8; ++j) hreg[j] = 0u;   // h_{-1} = 0

        const float* xrow = x + (size_t)(b0 + (lane & 15)) * kT * kD;

        unsigned xnext[8];
        load_x_tile(xrow, 0, xnext, lo);

        for (int t = 0; t < kT; ++t) {
            AB16 bx;
#pragma unroll
            for (int j = 0; j < 8; ++j) bx.u[j] = xnext[j];
            const int tn = (t + 1 < kT) ? (t + 1) : t;  // branchless prefetch
            load_x_tile(xrow, tn, xnext, lo);

            AB16 bh;
#pragma unroll
            for (int j = 0; j < 8; ++j) bh.u[j] = hreg[j];

            v8f acc0 = wmma_f32_f16(a_ih0, bx.v, cb0);  // S*(W_ih0 x_t^T + b)
            v8f acc1 = wmma_f32_f16(a_ih1, bx.v, cb1);
            acc0 = wmma_f32_f16(a_hh0, bh.v, acc0);     // + S*(W_hh0 h^T)
            acc1 = wmma_f32_f16(a_hh1, bh.v, acc1);

            float t0[8], t1[8];
            acc_to_B(acc0, acc1, t0, t1, hreg, lane);   // tanh + repack

#pragma unroll
            for (int j = 0; j < 8; ++j) hbuf[t & 1][lane][j] = hreg[j];
            __syncthreads();                            // publish h0_t
        }
    } else {
        // ============ layer-1 consumer wave + FC epilogue ============
        const v16h a_ih0 = load_A_tile(W_ih1,  0, 32, 32, lane, kS);
        const v16h a_ih1 = load_A_tile(W_ih1, 16, 32, 32, lane, kS);
        const v16h a_hh0 = load_A_tile(W_hh1,  0, 32, 32, lane, kS);
        const v16h a_hh1 = load_A_tile(W_hh1, 16, 32, 32, lane, kS);

        v8f cb0, cb1;
        float wf0[8], wf1[8];
#pragma unroll
        for (int v = 0; v < 8; ++v) {
            const int m = v + moff;
            cb0[v] = kS * (b_ih1[m]      + b_hh1[m]);
            cb1[v] = kS * (b_ih1[m + 16] + b_hh1[m + 16]);
            wf0[v] = W_fc[m];
            wf1[v] = W_fc[m + 16];
        }

        unsigned hreg[8];
#pragma unroll
        for (int j = 0; j < 8; ++j) hreg[j] = 0u;

        float t0[8], t1[8];

        for (int t = 0; t < kT; ++t) {
            __syncthreads();                            // wait for h0_t
            AB16 bx;
#pragma unroll
            for (int j = 0; j < 8; ++j) bx.u[j] = hbuf[t & 1][lane][j];

            AB16 bh;
#pragma unroll
            for (int j = 0; j < 8; ++j) bh.u[j] = hreg[j];

            v8f acc0 = wmma_f32_f16(a_ih0, bx.v, cb0);  // S*(W_ih1 h0^T + b)
            v8f acc1 = wmma_f32_f16(a_ih1, bx.v, cb1);
            acc0 = wmma_f32_f16(a_hh0, bh.v, acc0);     // + S*(W_hh1 h1^T)
            acc1 = wmma_f32_f16(a_hh1, bh.v, acc1);

            acc_to_B(acc0, acc1, t0, t1, hreg, lane);
        }

        // out[b0+n] = sum_m h1_last[m][n] * W_fc[m] + b_fc
        float partial = 0.0f;
#pragma unroll
        for (int v = 0; v < 8; ++v) partial += t0[v] * wf0[v] + t1[v] * wf1[v];
        partial += __shfl_xor(partial, 16, 32);         // combine lane halves
        if (lo) out[b0 + lane] = partial + b_fc[0];
    }
}

extern "C" void kernel_launch(void* const* d_in, const int* in_sizes, int n_in,
                              void* d_out, int out_size, void* d_ws, size_t ws_size,
                              hipStream_t stream) {
    (void)in_sizes; (void)n_in; (void)d_ws; (void)ws_size; (void)out_size;
    const float* x     = (const float*)d_in[0];
    const float* W_ih0 = (const float*)d_in[1];
    const float* W_hh0 = (const float*)d_in[2];
    const float* b_ih0 = (const float*)d_in[3];
    const float* b_hh0 = (const float*)d_in[4];
    const float* W_ih1 = (const float*)d_in[5];
    const float* W_hh1 = (const float*)d_in[6];
    const float* b_ih1 = (const float*)d_in[7];
    const float* b_hh1 = (const float*)d_in[8];
    const float* W_fc  = (const float*)d_in[9];
    const float* b_fc  = (const float*)d_in[10];
    float* out = (float*)d_out;

    dim3 grid(2048 / 16);   // one 16-row batch tile per workgroup
    dim3 block(64);         // 2 waves: layer-0 producer, layer-1 consumer
    hipLaunchKernelGGL(rnn2_wmma_fused, grid, block, 0, stream,
                       x, W_ih0, W_hh0, b_ih0, b_hh0,
                       W_ih1, W_hh1, b_ih1, b_hh1, W_fc, b_fc, out);
}
